// NonLinearPressureOperator_25056839205976
// MI455X (gfx1250) — matrix operand order repaired and checked by reading, not compiled, not run on hardware
//
#include <hip/hip_runtime.h>
#include <hip/hip_fp16.h>
#include <math.h>

#define HH   2048
#define WW   2048
#define NPIX (HH*WW)
#define WP2  2050                 // padded width/height (1-pixel zero halo)
#define NP2TOT (WP2*WP2)

typedef __attribute__((ext_vector_type(16))) _Float16 v16h;
typedef __attribute__((ext_vector_type(8)))  _Float16 v8h;
typedef __attribute__((ext_vector_type(8)))  float    v8f;

// ---------------------------------------------------------------------------
// 1) Global mean/std reduction: sum & sumsq for rhs (ch0) and mask (ch1)
// ---------------------------------------------------------------------------
__global__ __launch_bounds__(256) void stats_kernel(
    const float* __restrict__ rhs, const float* __restrict__ mask,
    float* __restrict__ stats)
{
    float s0 = 0.f, s1 = 0.f, s2 = 0.f, s3 = 0.f;
    for (size_t i = (size_t)blockIdx.x * blockDim.x + threadIdx.x;
         i < (size_t)NPIX; i += (size_t)gridDim.x * blockDim.x) {
        float a = rhs[i], b = mask[i];
        s0 += a; s1 += a * a; s2 += b; s3 += b * b;
    }
    __shared__ float sh[4][256];
    int tid = threadIdx.x;
    sh[0][tid] = s0; sh[1][tid] = s1; sh[2][tid] = s2; sh[3][tid] = s3;
    __syncthreads();
    for (int off = 128; off > 0; off >>= 1) {
        if (tid < off) {
            sh[0][tid] += sh[0][tid + off];
            sh[1][tid] += sh[1][tid + off];
            sh[2][tid] += sh[2][tid + off];
            sh[3][tid] += sh[3][tid + off];
        }
        __syncthreads();
    }
    if (tid == 0) {
        atomicAdd(&stats[0], sh[0][0]);
        atomicAdd(&stats[1], sh[1][0]);
        atomicAdd(&stats[2], sh[2][0]);
        atomicAdd(&stats[3], sh[3][0]);
    }
}

// ---------------------------------------------------------------------------
// 2) Repack w2 (16,16,3,3) f32 -> fp16 [outC][tap(0..9)][inC], tap 9 = zeros
// ---------------------------------------------------------------------------
__global__ void repack_w2_kernel(const float* __restrict__ w2,
                                 _Float16* __restrict__ w2h)
{
    int idx = blockIdx.x * blockDim.x + threadIdx.x;   // 16*10*16 = 2560
    if (idx >= 16 * 10 * 16) return;
    int i   = idx % 16;
    int tap = (idx / 16) % 10;
    int o   = idx / 160;
    float v = (tap < 9) ? w2[(size_t)o * 144 + i * 9 + tap] : 0.0f;
    w2h[idx] = (_Float16)v;
}

// ---------------------------------------------------------------------------
// 3) Fused standardize + conv1 (2->16) + bias + ReLU over the PADDED grid.
//    Border pixels of the 2050x2050 NHWC fp16 buffer are written as zeros,
//    so conv2 needs no bounds checks (EXEC stays all-ones for WMMA).
// ---------------------------------------------------------------------------
__global__ __launch_bounds__(256) void conv1_kernel(
    const float* __restrict__ rhs, const float* __restrict__ mask,
    const float* __restrict__ w1, const float* __restrict__ b1,
    const float* __restrict__ stats, _Float16* __restrict__ x1)
{
    __shared__ float sw[16 * 18];
    __shared__ float sb[16];
    for (int t = threadIdx.x; t < 288; t += blockDim.x) sw[t] = w1[t];
    if (threadIdx.x < 16) sb[threadIdx.x] = b1[threadIdx.x];
    __syncthreads();

    int pix = blockIdx.x * blockDim.x + threadIdx.x;
    if (pix >= NP2TOT) return;
    int yp = pix / WP2, xp = pix % WP2;
    size_t base = (size_t)pix * 16;

    if (yp == 0 || yp == WP2 - 1 || xp == 0 || xp == WP2 - 1) {
        v16h z = {};
        *(v16h*)(x1 + base) = z;     // zero halo
        return;
    }
    int y = yp - 1, x = xp - 1;

    const float n = (float)NPIX;
    float mu0 = stats[0] / n, mu1 = stats[2] / n;
    float is0 = 1.0f / (sqrtf(fmaxf(stats[1] / n - mu0 * mu0, 0.f)) + 1e-8f);
    float is1 = 1.0f / (sqrtf(fmaxf(stats[3] / n - mu1 * mu1, 0.f)) + 1e-8f);

    float v[18];
#pragma unroll
    for (int t = 0; t < 9; ++t) {
        int yy = y + t / 3 - 1, xx = x + t % 3 - 1;
        bool in = (yy >= 0 && yy < HH && xx >= 0 && xx < WW);
        size_t q = (size_t)yy * WW + xx;
        v[t]     = in ? (rhs[q]  - mu0) * is0 : 0.f;
        v[9 + t] = in ? (mask[q] - mu1) * is1 : 0.f;
    }

    v16h outv;
#pragma unroll
    for (int o = 0; o < 16; ++o) {
        float acc = sb[o];
#pragma unroll
        for (int t = 0; t < 9; ++t) {
            acc = fmaf(sw[o * 18 + t],     v[t],     acc);
            acc = fmaf(sw[o * 18 + 9 + t], v[9 + t], acc);
        }
        outv[o] = (_Float16)fmaxf(acc, 0.0f);
    }
    *(v16h*)(x1 + base) = outv;
}

// ---------------------------------------------------------------------------
// 4) Zero the 1-pixel halo frame of the padded x2 buffer (8196 pixels)
// ---------------------------------------------------------------------------
__global__ void zero_halo_kernel(_Float16* __restrict__ x2)
{
    int i = blockIdx.x * blockDim.x + threadIdx.x;
    int yp, xp;
    if (i < WP2)                { yp = 0;             xp = i; }
    else if (i < 2 * WP2)       { yp = WP2 - 1;       xp = i - WP2; }
    else if (i < 2 * WP2 + HH)  { yp = i - 2*WP2 + 1; xp = 0; }
    else if (i < 2 * WP2 + 2*HH){ yp = i - 2*WP2 - HH + 1; xp = WP2 - 1; }
    else return;
    v16h z = {};
    *(v16h*)(x2 + ((size_t)yp * WP2 + xp) * 16) = z;
}

// ---------------------------------------------------------------------------
// 5) conv2 (16->16, 3x3) via WMMA implicit GEMM, 4 tiles (64 px) per wave.
//    Per tap-pair: one A fragment (weights) reused by 4 WMMAs; B fragment per
//    lane = one padded-NHWC pixel's contiguous 32B channel vector. No bounds
//    checks anywhere -> no exec-mask branches around the WMMA stream.
// ---------------------------------------------------------------------------
__global__ __launch_bounds__(256) void conv2_wmma_kernel(
    const _Float16* __restrict__ x1, const _Float16* __restrict__ w2h,
    const float* __restrict__ b2, _Float16* __restrict__ x2)
{
    const int lane = threadIdx.x & 31;
    const int wg   = blockIdx.x * 8 + (threadIdx.x >> 5);   // global wave id
    const int y    = wg >> 5;                                // 32 x-tiles/row
    const int x0   = (wg & 31) << 6;                         // 64 px per wave

    const int col = lane & 15;   // pixel within 16-tile / weight row M
    const int sel = lane >> 4;   // 0: tapA (K 0..15) / out ch 0..7, 1: tapB (K 16..31) / out ch 8..15

    v8f acc0 = {}, acc1 = {}, acc2 = {}, acc3 = {};

#pragma unroll
    for (int p = 0; p < 5; ++p) {
        const int tapA  = 2 * p;
        const int tapB  = 2 * p + 1;                 // 9 at p=4 (zero weights)
        const int tapBc = (tapB > 8) ? 8 : tapB;     // clamped for addressing
        // A fragment: lane half sel holds all 16 channels of tap (sel ? tapB : tapA)
        union { v16h v; v8h h[2]; } a;
        a.h[0] = *(const v8h*)(w2h + (size_t)col * 160 + (sel ? tapB : tapA) * 16);
        a.h[1] = *(const v8h*)(w2h + (size_t)col * 160 + (sel ? tapB : tapA) * 16 + 8);

        const int dyA = tapA / 3 - 1,  dxA = tapA % 3 - 1;
        const int dyB = tapBc / 3 - 1, dxB = tapBc % 3 - 1;
        const int dy  = sel ? dyB : dyA;
        const int dx  = sel ? dxB : dxA;
        const size_t rowb = (size_t)(y + dy + 1) * WP2;
        const int    xb   = x0 + col + dx + 1;

        v16h b0 = *(const v16h*)(x1 + (rowb + xb)      * 16);
        v16h b1v = *(const v16h*)(x1 + (rowb + xb + 16) * 16);
        v16h b2v = *(const v16h*)(x1 + (rowb + xb + 32) * 16);
        v16h b3v = *(const v16h*)(x1 + (rowb + xb + 48) * 16);

        acc0 = __builtin_amdgcn_wmma_f32_16x16x32_f16(false, a.v, false, b0,  (short)0, acc0, false, false);
        acc1 = __builtin_amdgcn_wmma_f32_16x16x32_f16(false, a.v, false, b1v, (short)0, acc1, false, false);
        acc2 = __builtin_amdgcn_wmma_f32_16x16x32_f16(false, a.v, false, b2v, (short)0, acc2, false, false);
        acc3 = __builtin_amdgcn_wmma_f32_16x16x32_f16(false, a.v, false, b3v, (short)0, acc3, false, false);
    }

    // Epilogue. D layout: lane n<16, VGPR r = D[r][n]; lane 16+n = D[8+r][n]
    const int cb = sel * 8;
    const v8f bias = *(const v8f*)(b2 + cb);
    v8f accs[4] = {acc0, acc1, acc2, acc3};
#pragma unroll
    for (int t = 0; t < 4; ++t) {
        const size_t pi =
            ((size_t)(y + 1) * WP2 + (x0 + t * 16 + col + 1)) * 16 + cb;
        v8h res = *(const v8h*)(x1 + pi);
        v8h outv;
#pragma unroll
        for (int r = 0; r < 8; ++r) {
            float v = accs[t][r] + bias[r] + (float)res[r];
            outv[r] = (_Float16)fmaxf(v, 0.0f);
        }
        *(v8h*)(x2 + pi) = outv;
    }
}

// ---------------------------------------------------------------------------
// 6) conv3 (16->1, 3x3) + bias -> p (f32); reads zero-padded x2, no checks
// ---------------------------------------------------------------------------
__global__ __launch_bounds__(256) void conv3_kernel(
    const _Float16* __restrict__ x2, const float* __restrict__ w3,
    const float* __restrict__ b3, float* __restrict__ p)
{
    __shared__ float sw[144];
    for (int t = threadIdx.x; t < 144; t += blockDim.x) sw[t] = w3[t];
    __syncthreads();

    int pix = blockIdx.x * blockDim.x + threadIdx.x;
    if (pix >= NPIX) return;
    int y = pix >> 11, x = pix & (WW - 1);

    float acc = b3[0];
#pragma unroll
    for (int t = 0; t < 9; ++t) {
        int yy = y + t / 3, xx = x + t % 3;          // padded coords
        const v16h q = *(const v16h*)(x2 + ((size_t)yy * WP2 + xx) * 16);
#pragma unroll
        for (int i = 0; i < 16; ++i)
            acc = fmaf(sw[i * 9 + t], (float)q[i], acc);
    }
    p[pix] = acc;
}

// ---------------------------------------------------------------------------
// 7) One 3x3 box-smooth iteration with reflect padding
// ---------------------------------------------------------------------------
__global__ __launch_bounds__(256) void smooth_kernel(
    const float* __restrict__ in, float* __restrict__ out)
{
    int pix = blockIdx.x * blockDim.x + threadIdx.x;
    if (pix >= NPIX) return;
    int y = pix >> 11, x = pix & (WW - 1);
    float acc = 0.f;
#pragma unroll
    for (int dy = -1; dy <= 1; ++dy) {
        int yy = y + dy;
        yy = (yy < 0) ? -yy : ((yy >= HH) ? 2 * HH - 2 - yy : yy);
#pragma unroll
        for (int dx = -1; dx <= 1; ++dx) {
            int xx = x + dx;
            xx = (xx < 0) ? -xx : ((xx >= WW) ? 2 * WW - 2 - xx : xx);
            acc += in[(size_t)yy * WW + xx];
        }
    }
    out[pix] = acc * (1.0f / 9.0f);
}

// ---------------------------------------------------------------------------
// 8) clip(-100,100) * mask, nan_to_num -> output
// ---------------------------------------------------------------------------
__global__ __launch_bounds__(256) void finalize_kernel(
    const float* __restrict__ p, const float* __restrict__ mask,
    float* __restrict__ out)
{
    int pix = blockIdx.x * blockDim.x + threadIdx.x;
    if (pix >= NPIX) return;
    float v = fminf(fmaxf(p[pix], -100.0f), 100.0f);
    v *= mask[pix];
    if (!__builtin_isfinite(v)) v = 0.0f;
    out[pix] = v;
}

// ---------------------------------------------------------------------------
extern "C" void kernel_launch(void* const* d_in, const int* in_sizes, int n_in,
                              void* d_out, int out_size, void* d_ws, size_t ws_size,
                              hipStream_t stream)
{
    (void)in_sizes; (void)n_in; (void)out_size; (void)ws_size;

    const float* rhs  = (const float*)d_in[0];
    const float* mask = (const float*)d_in[1];
    const float* w1   = (const float*)d_in[2];
    const float* b1   = (const float*)d_in[3];
    const float* w2   = (const float*)d_in[4];
    const float* b2   = (const float*)d_in[5];
    const float* w3   = (const float*)d_in[6];
    const float* b3   = (const float*)d_in[7];
    float* out = (float*)d_out;

    // Workspace layout (~302 MB):
    //   [0,64)     stats; [256,5376) w2h fp16 [16][10][16]
    //   [8192,..)  x1 padded NHWC fp16 (2050*2050*16)  ~128.2 MB
    //   next       x2 padded NHWC fp16                 ~128.2 MB
    //   next       p0, p1 f32 ping-pong (16 MB each)
    char* ws = (char*)d_ws;
    float*    stats = (float*)(ws);
    _Float16* w2h   = (_Float16*)(ws + 256);
    _Float16* x1    = (_Float16*)(ws + 8192);
    _Float16* x2    = x1 + (size_t)NP2TOT * 16;
    float*    p0    = (float*)(x2 + (size_t)NP2TOT * 16);
    float*    p1    = p0 + NPIX;

    hipMemsetAsync(stats, 0, 4 * sizeof(float), stream);

    stats_kernel<<<1024, 256, 0, stream>>>(rhs, mask, stats);
    repack_w2_kernel<<<10, 256, 0, stream>>>(w2, w2h);
    zero_halo_kernel<<<(2 * WP2 + 2 * HH + 255) / 256, 256, 0, stream>>>(x2);
    conv1_kernel<<<(NP2TOT + 255) / 256, 256, 0, stream>>>(rhs, mask, w1, b1, stats, x1);
    conv2_wmma_kernel<<<(HH * 32) / 8, 256, 0, stream>>>(x1, w2h, b2, x2);
    conv3_kernel<<<NPIX / 256, 256, 0, stream>>>(x2, w3, b3, p0);
    for (int i = 0; i < 8; ++i) {
        const float* s = (i & 1) ? p1 : p0;
        float*       d = (i & 1) ? p0 : p1;
        smooth_kernel<<<NPIX / 256, 256, 0, stream>>>(s, d);
    }
    finalize_kernel<<<NPIX / 256, 256, 0, stream>>>(p0, mask, out);
}
